// GNNModel_19713899889202
// MI455X (gfx1250) — matrix operand order repaired and checked by reading, not compiled, not run on hardware
//
#include <hip/hip_runtime.h>
#include <hip/hip_bf16.h>

typedef __attribute__((ext_vector_type(16))) _Float16 v16h;
typedef __attribute__((ext_vector_type(8)))  float    v8f;

// ---------------------------------------------------------------------------
// Hardware FP32 atomic add (global_atomic_add_f32, no CAS loop)
// ---------------------------------------------------------------------------
__device__ __forceinline__ void atomAddF(float* p, float v) {
    unsafeAtomicAdd(p, v);
}

// ---------------------------------------------------------------------------
// Utility: zero a float buffer (grid-stride)
// ---------------------------------------------------------------------------
__global__ void zero_f32(float* __restrict__ p, long long n) {
    long long i = (long long)blockIdx.x * blockDim.x + threadIdx.x;
    long long stride = (long long)gridDim.x * blockDim.x;
    for (; i < n; i += stride) p[i] = 0.0f;
}

// ---------------------------------------------------------------------------
// Degree accumulation: one thread per edge, two fp32 atomics (L2-resident)
// ---------------------------------------------------------------------------
__global__ void degree_kernel(const int* __restrict__ src, const int* __restrict__ dst,
                              float* __restrict__ deg_out, float* __restrict__ deg_in,
                              int n_edges) {
    int e = blockIdx.x * blockDim.x + threadIdx.x;
    if (e >= n_edges) return;
    atomAddF(&deg_out[src[e]], 1.0f);
    atomAddF(&deg_in[dst[e]], 1.0f);
}

// In-place: d = rsqrt(max(d, 1))
__global__ void norm_kernel(float* __restrict__ d, long long n) {
    long long i = (long long)blockIdx.x * blockDim.x + threadIdx.x;
    long long stride = (long long)gridDim.x * blockDim.x;
    for (; i < n; i += stride) d[i] = rsqrtf(fmaxf(d[i], 1.0f));
}

// ---------------------------------------------------------------------------
// Edge scatter: agg[dst[e], :] += x[src[e], :] * (scale ? scale[src[e]] : 1)
// F/4 threads per edge, float4 loads, 4 fp32 atomics each.
// x (<=12.8MB) and agg (<=12.8MB) both live in the 192MB L2 -> L2-rate atomics.
// ---------------------------------------------------------------------------
template <int F>
__global__ void scatter_kernel(const float* __restrict__ x,
                               const int*   __restrict__ src,
                               const int*   __restrict__ dst,
                               const float* __restrict__ scale,  // may be null
                               float*       __restrict__ agg,
                               int n_edges) {
    constexpr int TPE = F / 4;                  // threads per edge
    long long t = (long long)blockIdx.x * blockDim.x + threadIdx.x;
    long long e = t / TPE;
    int f4 = (int)(t % TPE);
    if (e >= n_edges) return;

    int s = src[e];
    int d = dst[e];
    float sc = scale ? scale[s] : 1.0f;

    const float4 v = *(const float4*)(x + (size_t)s * F + f4 * 4);
    float* o = agg + (size_t)d * F + f4 * 4;
    atomAddF(o + 0, v.x * sc);
    atomAddF(o + 1, v.y * sc);
    atomAddF(o + 2, v.z * sc);
    atomAddF(o + 3, v.w * sc);
}

// ---------------------------------------------------------------------------
// WMMA GEMM: out[N, OUT] = op( (in * rscale) @ W + bias )
//   in:  [n_nodes, 64] f32, rows optionally scaled by rscale[row]
//   W:   [64, OUT] f32 row-major
//   One wave per 16-row tile; K=64 -> two v_wmma_f32_16x16x32_f16 per n-tile.
//   f16 inputs, f32 accumulate (CDNA5 WMMA path; no MFMA on gfx1250).
//
// VGPR layouts (CDNA5 ISA):
//   A 16x32 f16: lane L holds row M=L&15; lanes<16 -> K {kb..kb+15}
//                with kb=0, lanes>=16 with kb=16 (contiguous k, not the doc's order).
//   B 32x16 f16: lane L holds col N=L&15; lanes<16 -> K 0..15, lanes>=16 -> K 16..31 (same k order as A here).
//   C/D 16x16 f32: VGPR r: lanes<16 -> (M=r, N=lane), lanes>=16 -> (M=r+8, N=lane-16).
// ---------------------------------------------------------------------------
template <int OUT, bool RELU>
__global__ void gemm_wmma_kernel(const float* __restrict__ in,
                                 const float* __restrict__ rscale,  // may be null
                                 const float* __restrict__ W,
                                 const float* __restrict__ bias,    // may be null
                                 float* __restrict__ out,
                                 int n_tiles) {
    const int wave = blockIdx.x * (blockDim.x >> 5) + (threadIdx.x >> 5);
    const int lane = threadIdx.x & 31;
    if (wave >= n_tiles) return;

    const int  mrow   = lane & 15;
    const bool hiHalf = (lane >= 16);
    const int  row    = wave * 16 + mrow;

    // ---- Pack A tile (16 rows x K=64) as two 16x32 f16 operands ----
    const float* arow = in + (size_t)row * 64;
    const float  rs   = rscale ? rscale[row] : 1.0f;
    const int    kb   = hiHalf ? 16 : 0;

    v16h a0, a1;
#pragma unroll
    for (int j = 0; j < 8; ++j) {
        a0[j]     = (_Float16)(arow[kb + j]           * rs);
        a0[j + 8] = (_Float16)(arow[kb + 8 + j]       * rs);
        a1[j]     = (_Float16)(arow[32 + kb + j]      * rs);
        a1[j + 8] = (_Float16)(arow[32 + kb + 8 + j]  * rs);
    }

    const int ncol_lane = lane & 15;
    const int kbb       = hiHalf ? 16 : 0;

#pragma unroll
    for (int nt = 0; nt < OUT / 16; ++nt) {
        const int ncol = nt * 16 + ncol_lane;

        // ---- Pack B tiles (32x16 f16 each), W is [64, OUT] row-major ----
        v16h b0, b1;
#pragma unroll
        for (int j = 0; j < 16; ++j) {
            b0[j] = (_Float16)W[(kbb + j) * OUT + ncol];
            b1[j] = (_Float16)W[(32 + kbb + j) * OUT + ncol];
        }

        v8f acc = {};
        acc = __builtin_amdgcn_wmma_f32_16x16x32_f16(
            false, a0, false, b0, (short)0, acc, false, false);
        acc = __builtin_amdgcn_wmma_f32_16x16x32_f16(
            false, a1, false, b1, (short)0, acc, false, false);

        const float bv = bias ? bias[ncol] : 0.0f;
#pragma unroll
        for (int r = 0; r < 8; ++r) {
            const int mm = r + (hiHalf ? 8 : 0);
            float v = acc[r] + bv;
            if (RELU) v = fmaxf(v, 0.0f);
            out[(size_t)(wave * 16 + mm) * OUT + ncol] = v;
        }
    }
}

// ---------------------------------------------------------------------------
// Finalize: out[i, c] = out[i, c] * norm_dst[i] + b2[c]   (OUT = 32)
// ---------------------------------------------------------------------------
__global__ void finalize_kernel(float* __restrict__ out,
                                const float* __restrict__ norm_dst,
                                const float* __restrict__ b2,
                                long long total) {
    long long t = (long long)blockIdx.x * blockDim.x + threadIdx.x;
    if (t >= total) return;
    long long i = t >> 5;   // / 32
    int c = (int)(t & 31);
    out[t] = out[t] * norm_dst[i] + b2[c];
}

// ---------------------------------------------------------------------------
// Launch
// ---------------------------------------------------------------------------
extern "C" void kernel_launch(void* const* d_in, const int* in_sizes, int n_in,
                              void* d_out, int out_size, void* d_ws, size_t ws_size,
                              hipStream_t stream) {
    const float* features = (const float*)d_in[0];  // [N, 64]
    const int*   src      = (const int*)  d_in[1];  // [E]
    const int*   dst      = (const int*)  d_in[2];  // [E]
    const float* W1       = (const float*)d_in[3];  // [64, 64]
    const float* b1       = (const float*)d_in[4];  // [64]
    const float* W2       = (const float*)d_in[5];  // [64, 32]
    const float* b2       = (const float*)d_in[6];  // [32]
    float*       out      = (float*)d_out;          // [N, 32]

    const int IN_FEATS = 64, HIDDEN = 64, NUM_CLASSES = 32;
    const int N = in_sizes[0] / IN_FEATS;           // 50000
    const int E = in_sizes[1];                      // 800000

    // Workspace layout (floats)
    float* ws       = (float*)d_ws;
    float* norm_src = ws;                             // [N] out-degree -> rsqrt norm
    float* norm_dst = ws + N;                         // [N] in-degree  -> rsqrt norm
    float* agg1     = ws + 2 * (size_t)N;             // [N, 64]; reused as p2 [N, 32]
    float* h1       = agg1 + (size_t)N * HIDDEN;      // [N, 64]
    float* p2       = agg1;                           // alias (agg1 dead after GEMM1)

    const int TB = 256;
    auto blocks = [&](long long work) { return (int)((work + TB - 1) / TB); };

    // ---- Degrees + norms ----
    zero_f32<<<blocks(2LL * N), TB, 0, stream>>>(norm_src, 2LL * N);
    degree_kernel<<<blocks(E), TB, 0, stream>>>(src, dst, norm_src, norm_dst, E);
    norm_kernel<<<blocks(2LL * N), TB, 0, stream>>>(norm_src, 2LL * N);

    // ---- Layer 1: agg1 = segsum((x * norm_src)[src] -> dst) ----
    zero_f32<<<blocks((long long)N * IN_FEATS), TB, 0, stream>>>(agg1, (long long)N * IN_FEATS);
    scatter_kernel<64><<<blocks((long long)E * (64 / 4)), TB, 0, stream>>>(
        features, src, dst, norm_src, agg1, E);

    // h1 = relu((agg1 * norm_dst) @ W1 + b1)   [WMMA]
    const int tiles = N / 16;                        // 3125 (N % 16 == 0)
    const int waves_per_block = TB / 32;
    gemm_wmma_kernel<64, true><<<(tiles + waves_per_block - 1) / waves_per_block, TB, 0, stream>>>(
        agg1, norm_dst, W1, b1, h1, tiles);

    // ---- Layer 2 (GEMM commuted before aggregation: segsum is linear) ----
    // p2 = (h1 * norm_src) @ W2     [N, 32]   [WMMA]
    gemm_wmma_kernel<32, false><<<(tiles + waves_per_block - 1) / waves_per_block, TB, 0, stream>>>(
        h1, norm_src, W2, nullptr, p2, tiles);

    // out = segsum(p2[src] -> dst)  (only 32 feats/edge: half the edge traffic)
    zero_f32<<<blocks((long long)N * NUM_CLASSES), TB, 0, stream>>>(out, (long long)N * NUM_CLASSES);
    scatter_kernel<32><<<blocks((long long)E * (32 / 4)), TB, 0, stream>>>(
        p2, src, dst, nullptr, out, E);

    // out = out * norm_dst + b2
    finalize_kernel<<<blocks((long long)N * NUM_CLASSES), TB, 0, stream>>>(
        out, norm_dst, b2, (long long)N * NUM_CLASSES);
}
